// GGMSA_27522150433350
// MI455X (gfx1250) — matrix operand order repaired and checked by reading, not compiled, not run on hardware
//
#include <hip/hip_runtime.h>
#include <math.h>

// ---------------------------------------------------------------------------
// Swin-V2 windowed attention block on MI455X (gfx1250, wave32, WMMA).
// Memory-bound (~77 GFLOP vs ~1.5 GB HBM traffic @ 23.3 TB/s): all GEMMs via
// v_wmma_f32_16x16x32_bf16 (fp32 accum), LDS tiles double-buffered, fragment
// data laid out in LDS so every fragment load is a contiguous ds_load_b128.
// ---------------------------------------------------------------------------

typedef __attribute__((ext_vector_type(16))) __bf16 v16bf;
typedef __attribute__((ext_vector_type(8)))  float  v8f;

#define HEADS   8
#define HD      32
#define LWIN    64
#define HIMG    126
#define WIMG    126
#define BATCH   8
#define C_IN    256
#define C3      768
#define MROWS   (BATCH * HIMG * WIMG)   // 127008
#define NWIN    (BATCH * 256)           // 2048 windows (16x16 per batch)
#define LOG100F 4.6051702f

// f32 -> bf16 (round-to-nearest-even) as raw bits
static __device__ __forceinline__ unsigned short f2bf(float f) {
  union { float f; unsigned u; } v; v.f = f;
  unsigned r = v.u + 0x7FFFu + ((v.u >> 16) & 1u);
  return (unsigned short)(r >> 16);
}
static __device__ __forceinline__ __bf16 bf_bits(unsigned short s) {
  union { unsigned short s; __bf16 b; } v; v.s = s; return v.b;
}

// ---------------------------------------------------------------------------
// Kernels 1/5: tiled WMMA GEMM  C[M,N] = A[M,K] * B[K,N]  (f32 in, bf16 MACs)
// Block = 128 threads (4 waves), 128x64 output tile. Each wave owns 2 M-sub-
// tiles x 4 N-subtiles = 8 WMMAs per K-step. LDS is double-buffered; the B
// tile is stored fragment-major so B-fragment loads are contiguous b128s.
// remap==1: A rows address the padded attention output with the (1,1) crop
// folded in (final projection GEMM).
// ---------------------------------------------------------------------------
__global__ __launch_bounds__(128) void gemm_bf16_wmma(
    const float* __restrict__ A, const float* __restrict__ B,
    float* __restrict__ C, int M, int N, int K, int remap)
{
  // A tile: row-major [128][32]  (A-fragment reads are contiguous per half)
  __shared__ unsigned short sA[2][128 * 32];
  // B tile swizzled: element (k, n) at [(n*2 + (k>>4))*16 + (k&15)]
  __shared__ unsigned short sB[2][64 * 32];

  const int tid  = threadIdx.x;
  const int lane = tid & 31;
  const int wave = tid >> 5;
  const int lm   = lane & 15;      // position within 16-lane group
  const int lh   = lane >> 4;      // which 16-lane half
  const int m0   = blockIdx.y * 128;
  const int n0   = blockIdx.x * 64;

  const v8f vzero = {0.f, 0.f, 0.f, 0.f, 0.f, 0.f, 0.f, 0.f};
  v8f acc[2][4];
#pragma unroll
  for (int mt = 0; mt < 2; mt++)
#pragma unroll
    for (int nt = 0; nt < 4; nt++) acc[mt][nt] = vzero;

  auto loadA = [&](int k0, int buf) {
    for (int idx = tid; idx < 128 * 32; idx += 128) {
      int mm = idx >> 5, kk = idx & 31;
      int row = m0 + mm;
      float v = 0.f;
      if (row < M) {
        long base;
        if (!remap) {
          base = (long)row * K;
        } else {
          // crop map: out row -> padded [B,128,128,C] pixel (y+1, x+1)
          int b  = row / 15876;
          int rm = row - b * 15876;
          int yy = rm / 126;
          int xx = rm - yy * 126;
          base = ((long)((b * 128 + yy + 1) * 128 + (xx + 1))) * K;
        }
        v = A[base + k0 + kk];
        if (k0 + 64 < K) __builtin_prefetch(&A[base + k0 + 64 + kk], 0, 1);
      }
      sA[buf][mm * 32 + kk] = f2bf(v);
    }
  };
  auto loadB = [&](int k0, int buf) {
    for (int idx = tid; idx < 32 * 64; idx += 128) {
      int kk = idx >> 6, nn = idx & 63;
      float v = B[(long)(k0 + kk) * N + n0 + nn];
      sB[buf][(nn * 2 + (kk >> 4)) * 16 + (kk & 15)] = f2bf(v);
    }
  };

  loadA(0, 0);
  loadB(0, 0);
  __syncthreads();

  for (int k0 = 0; k0 < K; k0 += 32) {
    const int buf = (k0 >> 5) & 1;
    if (k0 + 32 < K) {            // fetch next tile into the other buffer
      loadA(k0 + 32, buf ^ 1);
      loadB(k0 + 32, buf ^ 1);
    }

    // A fragments (16x32 bf16): contiguous k = lh*16 + i, matching B
    v16bf af[2];
#pragma unroll
    for (int mt = 0; mt < 2; mt++) {
#pragma unroll
      for (int i = 0; i < 16; i++) {
        af[mt][i] = bf_bits(
            sA[buf][((wave * 2 + mt) * 16 + lm) * 32 + lh * 16 + i]);
      }
    }
#pragma unroll
    for (int nt = 0; nt < 4; nt++) {
      // B fragment: 16 contiguous ushorts thanks to the swizzled store
      v16bf bfm;
      const int bbase = ((nt * 16 + lm) * 2 + lh) * 16;
#pragma unroll
      for (int i = 0; i < 16; i++) bfm[i] = bf_bits(sB[buf][bbase + i]);
#pragma unroll
      for (int mt = 0; mt < 2; mt++) {
        acc[mt][nt] = __builtin_amdgcn_wmma_f32_16x16x32_bf16(
            false, af[mt], false, bfm, (short)0, acc[mt][nt], false, false);
      }
    }
    __syncthreads();
  }

  // store (C/D layout: lane half selects M+8, VGPR j = M row). Streaming
  // outputs bypass cache rinse via non-temporal stores.
#pragma unroll
  for (int mt = 0; mt < 2; mt++) {
#pragma unroll
    for (int nt = 0; nt < 4; nt++) {
#pragma unroll
      for (int j = 0; j < 8; j++) {
        int row = m0 + (wave * 2 + mt) * 16 + lh * 8 + j;
        int col = n0 + nt * 16 + lm;
        if (row < M)
          __builtin_nontemporal_store(acc[mt][nt][j], &C[(long)row * N + col]);
      }
    }
  }
}

// ---------------------------------------------------------------------------
// Kernel 2a: CPB MLP over the 15x15 relative-position grid: 2 -> 512 -> 8
// ---------------------------------------------------------------------------
__global__ __launch_bounds__(256) void cpb_mlp(
    const float* __restrict__ w0, const float* __restrict__ b0,
    const float* __restrict__ w1, float* __restrict__ tmp /*225x8*/)
{
  __shared__ float part[256 * 8];
  int t   = threadIdx.x;
  int rel = blockIdx.x;                 // 0..224
  int iy  = rel / 15, ix = rel % 15;
  float ry = (float)(iy - 7) * (8.0f / 7.0f);
  float rx = (float)(ix - 7) * (8.0f / 7.0f);
  const float il8 = 0.48089834696298783f;   // 1/ln(8)
  float sy = (ry > 0.f) ? 1.f : ((ry < 0.f) ? -1.f : 0.f);
  float sx = (rx > 0.f) ? 1.f : ((rx < 0.f) ? -1.f : 0.f);
  ry = sy * log1pf(fabsf(ry)) * il8;
  rx = sx * log1pf(fabsf(rx)) * il8;

  float h0 = fmaxf(ry * w0[t]       + rx * w0[512 + t]       + b0[t],       0.f);
  float h1 = fmaxf(ry * w0[t + 256] + rx * w0[512 + t + 256] + b0[t + 256], 0.f);
#pragma unroll
  for (int hh = 0; hh < 8; hh++)
    part[t * 8 + hh] = h0 * w1[t * 8 + hh] + h1 * w1[(t + 256) * 8 + hh];
  __syncthreads();
  for (int st = 128; st >= 1; st >>= 1) {
    if (t < st) {
#pragma unroll
      for (int hh = 0; hh < 8; hh++) part[t * 8 + hh] += part[(t + st) * 8 + hh];
    }
    __syncthreads();
  }
  if (t < 8) tmp[rel * 8 + t] = 16.f / (1.f + __expf(-part[t]));
}

// Kernel 2b: scatter to biasT[8][64][64]
__global__ __launch_bounds__(256) void cpb_scatter(
    const float* __restrict__ tmp, float* __restrict__ biasT)
{
  int idx = blockIdx.x * 256 + threadIdx.x;
  if (idx >= 8 * 64 * 64) return;
  int hh = idx >> 12;
  int q  = (idx >> 6) & 63, k = idx & 63;
  int d0 = (q >> 3) - (k >> 3) + 7;
  int d1 = (q & 7)  - (k & 7)  + 7;
  biasT[idx] = tmp[(d0 * 15 + d1) * 8 + hh];
}

// ---------------------------------------------------------------------------
// Kernel 3: fused 3x3 depthwise conv + LayerNorm(768) + q/v bias + window
// partition. One wave32 per padded pixel (24 channels/lane), LN reduction via
// __shfl_xor. Writes windowed q/k/v [2048, 8, 64, 32]; pad pixels -> zeros.
// ---------------------------------------------------------------------------
__global__ __launch_bounds__(256) void dwln_window(
    const float* __restrict__ qkv, const float* __restrict__ wdw,
    const float* __restrict__ gamma, const float* __restrict__ beta,
    const float* __restrict__ qb, const float* __restrict__ vb,
    float* __restrict__ qo, float* __restrict__ ko, float* __restrict__ vo)
{
  int wave = threadIdx.x >> 5, lane = threadIdx.x & 31;
  int p = blockIdx.x * 8 + wave;            // padded pixel id, < 8*128*128
  int b   = p >> 14;
  int rem = p & 16383;
  int py  = rem >> 7, px = rem & 127;       // padded coords
  int y   = py - 1,  x  = px - 1;           // image coords (hb=wb=1)
  bool inimg = (y >= 0 && y < HIMG && x >= 0 && x < WIMG);

  float t[24];
  float mu = 0.f, inv = 0.f;
  if (inimg) {
    float s = 0.f, s2 = 0.f;
#pragma unroll
    for (int j = 0; j < 24; j++) {
      int c = lane + j * 32;
      float a = 0.f;
#pragma unroll
      for (int ky = 0; ky < 3; ky++) {
        int yy = y + ky - 1;
        if (yy < 0 || yy >= HIMG) continue;
#pragma unroll
        for (int kx = 0; kx < 3; kx++) {
          int xx = x + kx - 1;
          if (xx < 0 || xx >= WIMG) continue;
          a += qkv[((long)(b * HIMG + yy) * WIMG + xx) * C3 + c] *
               wdw[(ky * 3 + kx) * C3 + c];
        }
      }
      t[j] = a; s += a; s2 += a * a;
    }
    for (int m = 16; m >= 1; m >>= 1) {
      s  += __shfl_xor(s,  m, 32);
      s2 += __shfl_xor(s2, m, 32);
    }
    mu = s * (1.f / 768.f);
    float var = s2 * (1.f / 768.f) - mu * mu;
    inv = rsqrtf(var + 1e-5f);
  }

  int wy = py >> 3, wx = px >> 3;
  int n  = b * 256 + wy * 16 + wx;          // window id
  int r  = (py & 7) * 8 + (px & 7);         // row within window (L=64)
#pragma unroll
  for (int j = 0; j < 24; j++) {
    int c = lane + j * 32;
    float val = 0.f;
    if (inimg) {
      val = (t[j] - mu) * inv * gamma[c] + beta[c];
      if (c < 256)        val += qb[c];
      else if (c >= 512)  val += vb[c - 512];
    }
    int which = c >> 8;                     // 0=q 1=k 2=v
    int head  = (c & 255) >> 5;
    int d     = c & 31;
    float* dst = (which == 0) ? qo : ((which == 1) ? ko : vo);
    dst[(((long)(n * 8 + head)) * 64 + r) * 32 + d] = val;
  }
}

// ---------------------------------------------------------------------------
// Kernel 4: windowed cosine attention, one block (128 thr = 4 waves) per
// (window, head). l2-norm q/k -> bf16 LDS, S = q@k^T via WMMA, +CPB bias +
// pad mask, layout-aware in-register softmax, P@V via WMMA. V is stored
// TRANSPOSED in LDS so the P@V B-fragments are contiguous b128 loads.
// ---------------------------------------------------------------------------
__global__ __launch_bounds__(128) void window_attn(
    const float* __restrict__ qi, const float* __restrict__ ki,
    const float* __restrict__ vi, const float* __restrict__ scale,
    const float* __restrict__ biasT, float* __restrict__ obuf)
{
  __shared__ unsigned short sq[64 * 32];   // [row][k]
  __shared__ unsigned short sk[64 * 32];   // [key][k]   (B-frag contiguous)
  __shared__ unsigned short svT[32 * 64];  // [d][key]   (B-frag contiguous)
  __shared__ unsigned short sp[64 * 64];   // [row][key] (A-frag contiguous)

  const int tid  = threadIdx.x;
  const int lane = tid & 31, wave = tid >> 5;
  const int lm   = lane & 15, lh = lane >> 4;
  const int blk  = blockIdx.x;              // n*8 + h
  const int n    = blk >> 3, h = blk & 7;
  const long base = (long)blk * 64 * 32;

  float ls = __expf(fminf(scale[h], LOG100F));

  // --- normalize q (tid<64) / k (tid>=64): one row of 32 per thread ---
  {
    int r = tid & 63;
    const float* src = (tid < 64) ? (qi + base + r * 32) : (ki + base + r * 32);
    float tmp[32];
    float ss = 0.f;
#pragma unroll
    for (int d = 0; d < 32; d++) { float v = src[d]; tmp[d] = v; ss += v * v; }
    float f = rsqrtf(fmaxf(ss, 1.55e-5f));
    if (tid < 64) f *= ls;                  // fold logit scale into q
    unsigned short* dst = (tid < 64) ? sq : sk;
#pragma unroll
    for (int d = 0; d < 32; d++) dst[r * 32 + d] = f2bf(tmp[d] * f);
  }
  {
    int e0 = tid * 16;
#pragma unroll
    for (int i = 0; i < 16; i++) {
      int e = e0 + i;
      int kr = e >> 5, d = e & 31;
      svT[d * 64 + kr] = f2bf(vi[base + e]);   // transpose on the way in
    }
  }
  __syncthreads();

  // --- logits S[64,64] = q̂ @ k̂^T ---
  const v8f vzero = {0.f, 0.f, 0.f, 0.f, 0.f, 0.f, 0.f, 0.f};
  v8f acc[4];
#pragma unroll
  for (int nt = 0; nt < 4; nt++) acc[nt] = vzero;

  v16bf af;
#pragma unroll
  for (int i = 0; i < 16; i++) {
    af[i] = bf_bits(sq[(wave * 16 + lm) * 32 + lh * 16 + i]);
  }
#pragma unroll
  for (int nt = 0; nt < 4; nt++) {
    v16bf bfm;
    const int kb = (nt * 16 + lm) * 32 + lh * 16;   // contiguous 16 ushorts
#pragma unroll
    for (int i = 0; i < 16; i++) bfm[i] = bf_bits(sk[kb + i]);
    acc[nt] = __builtin_amdgcn_wmma_f32_16x16x32_bf16(
        false, af, false, bfm, (short)0, acc[nt], false, false);
  }

  // --- + CPB bias + padding mask ---
  int wy = (n >> 4) & 15, wx = n & 15;
  float cb[4];
#pragma unroll
  for (int nt = 0; nt < 4; nt++) {
    int col = nt * 16 + lm;
    int py = wy * 8 + (col >> 3), px = wx * 8 + (col & 7);
    bool valid = (py >= 1 && py <= 126 && px >= 1 && px <= 126);
    cb[nt] = valid ? 0.f : -100.f;
  }
#pragma unroll
  for (int nt = 0; nt < 4; nt++) {
#pragma unroll
    for (int j = 0; j < 8; j++) {
      int row = wave * 16 + lh * 8 + j;
      int col = nt * 16 + lm;
      acc[nt][j] += biasT[(h * 64 + row) * 64 + col] + cb[nt];
    }
  }

  // --- softmax per row (row = fixed VGPR j within a 16-lane group) ---
  float rs[8];
#pragma unroll
  for (int j = 0; j < 8; j++) {
    float m = fmaxf(fmaxf(acc[0][j], acc[1][j]), fmaxf(acc[2][j], acc[3][j]));
    for (int msk = 8; msk >= 1; msk >>= 1) m = fmaxf(m, __shfl_xor(m, msk, 32));
    float s = 0.f;
#pragma unroll
    for (int nt = 0; nt < 4; nt++) {
      float pexp = __expf(acc[nt][j] - m);
      acc[nt][j] = pexp;
      s += pexp;
    }
    for (int msk = 8; msk >= 1; msk >>= 1) s += __shfl_xor(s, msk, 32);
    rs[j] = 1.f / s;
  }
#pragma unroll
  for (int nt = 0; nt < 4; nt++) {
#pragma unroll
    for (int j = 0; j < 8; j++) {
      sp[(wave * 16 + lh * 8 + j) * 64 + nt * 16 + lm] = f2bf(acc[nt][j] * rs[j]);
    }
  }
  __syncthreads();

  // --- O[64,32] = P[64,64] @ V[64,32]  (two K=32 WMMA steps) ---
  v8f oacc[2];
  oacc[0] = vzero; oacc[1] = vzero;
#pragma unroll
  for (int kk = 0; kk < 64; kk += 32) {
    v16bf pa;
#pragma unroll
    for (int i = 0; i < 16; i++) {
      pa[i] = bf_bits(sp[(wave * 16 + lm) * 64 + kk + lh * 16 + i]);
    }
#pragma unroll
    for (int nt = 0; nt < 2; nt++) {
      v16bf vb_;
      const int vbase = (nt * 16 + lm) * 64 + kk + lh * 16;  // contiguous
#pragma unroll
      for (int i = 0; i < 16; i++) vb_[i] = bf_bits(svT[vbase + i]);
      oacc[nt] = __builtin_amdgcn_wmma_f32_16x16x32_bf16(
          false, pa, false, vb_, (short)0, oacc[nt], false, false);
    }
  }

  // --- scatter to padded spatial layout [B,128,128,256] ---
  int bb = n >> 8;
#pragma unroll
  for (int nt = 0; nt < 2; nt++) {
#pragma unroll
    for (int j = 0; j < 8; j++) {
      int r = wave * 16 + lh * 8 + j;
      int d = nt * 16 + lm;
      int py = wy * 8 + (r >> 3), px = wx * 8 + (r & 7);
      obuf[((long)(bb * 128 + py) * 128 + px) * 256 + h * 32 + d] = oacc[nt][j];
    }
  }
}

// ---------------------------------------------------------------------------
// Launch
// ---------------------------------------------------------------------------
extern "C" void kernel_launch(void* const* d_in, const int* in_sizes, int n_in,
                              void* d_out, int out_size, void* d_ws, size_t ws_size,
                              hipStream_t stream)
{
  (void)in_sizes; (void)n_in; (void)out_size; (void)ws_size;
  const float* x      = (const float*)d_in[0];
  const float* w_qkv  = (const float*)d_in[1];
  const float* w_dw   = (const float*)d_in[2];
  const float* gamma  = (const float*)d_in[3];
  const float* beta   = (const float*)d_in[4];
  const float* q_bias = (const float*)d_in[5];
  const float* v_bias = (const float*)d_in[6];
  const float* scale  = (const float*)d_in[7];
  const float* cpb_w0 = (const float*)d_in[8];
  const float* cpb_b0 = (const float*)d_in[9];
  const float* cpb_w1 = (const float*)d_in[10];
  const float* w_proj = (const float*)d_in[11];

  char* ws = (char*)d_ws;
  size_t off = 0;
  auto alloc = [&](size_t bytes) -> void* {
    void* p = ws + off;
    off += (bytes + 255) & ~(size_t)255;
    return p;
  };
  float* qkv_buf = (float*)alloc((size_t)MROWS * C3 * 4);            // 390 MB
  float* q_buf   = (float*)alloc((size_t)NWIN * 8 * 64 * 32 * 4);    // 134 MB
  float* k_buf   = (float*)alloc((size_t)NWIN * 8 * 64 * 32 * 4);
  float* v_buf   = (float*)alloc((size_t)NWIN * 8 * 64 * 32 * 4);
  float* o_buf   = (float*)alloc((size_t)8 * 128 * 128 * 256 * 4);   // 134 MB
  float* cpb_tmp = (float*)alloc((size_t)225 * 8 * 4);
  float* biasT   = (float*)alloc((size_t)8 * 64 * 64 * 4);

  // 1) QKV GEMM: [127008,256] x [256,768]
  {
    dim3 grid(C3 / 64, (MROWS + 127) / 128);
    gemm_bf16_wmma<<<grid, 128, 0, stream>>>(x, w_qkv, qkv_buf,
                                             MROWS, C3, C_IN, 0);
  }
  // 2) CPB bias table
  cpb_mlp<<<225, 256, 0, stream>>>(cpb_w0, cpb_b0, cpb_w1, cpb_tmp);
  cpb_scatter<<<(8 * 64 * 64 + 255) / 256, 256, 0, stream>>>(cpb_tmp, biasT);
  // 3) depthwise conv + LN + bias + window partition
  dwln_window<<<(8 * 128 * 128) / 8, 256, 0, stream>>>(
      qkv_buf, w_dw, gamma, beta, q_bias, v_bias, q_buf, k_buf, v_buf);
  // 4) windowed attention (2048 windows x 8 heads)
  window_attn<<<NWIN * HEADS, 128, 0, stream>>>(q_buf, k_buf, v_buf,
                                                scale, biasT, o_buf);
  // 5) projection GEMM with crop remap: [127008,256] x [256,256] -> d_out
  {
    dim3 grid(C_IN / 64, (MROWS + 127) / 128);
    gemm_bf16_wmma<<<grid, 128, 0, stream>>>(o_buf, w_proj, (float*)d_out,
                                             MROWS, C_IN, C_IN, 1);
  }
}
